// PACCS_46840913330689
// MI455X (gfx1250) — matrix operand order repaired and checked by reading, not compiled, not run on hardware
//
#include <hip/hip_runtime.h>
#include <cstdint>

#define N_NODES 10000
#define N_EDGES 50000
#define NPAD    10048      // 157 * 64
#define H       64
#define KSL     65         // 64 weight slices + 1 bias slice
#define MT      157        // NPAD / 64
#define SP      80         // padded LDS row stride (elements), 160B = 16B-aligned

typedef __bf16 bf16;
typedef __attribute__((ext_vector_type(16))) __bf16 v16bf;
typedef __attribute__((ext_vector_type(8)))  __bf16 bf16x8;
typedef __attribute__((ext_vector_type(8)))  float  v8f;

// ---------------------------------------------------------------- utility
__global__ void k_zero(float* __restrict__ p, int n) {
  int i = blockIdx.x * blockDim.x + threadIdx.x;
  if (i < n) p[i] = 0.f;
}

// h[n][o] = relu(x[n]@W0 + b0); pad rows -> 0
__global__ void k_lin0(const float* __restrict__ x, const float* __restrict__ w,
                       const float* __restrict__ b, float* __restrict__ h) {
  int n = blockIdx.x, o = threadIdx.x;
  float acc = 0.f;
  if (n < N_NODES) {
    acc = b[o];
    #pragma unroll
    for (int k = 0; k < 20; ++k) acc = fmaf(x[n * 20 + k], w[k * H + o], acc);
    acc = fmaxf(acc, 0.f);
  }
  h[(size_t)n * H + o] = acc;
}

// eh[e][o] = relu(edge_attr[e]@W1 + b1)
__global__ void k_nn1(const float* __restrict__ ea, const float* __restrict__ w,
                      const float* __restrict__ b, float* __restrict__ eh) {
  int e = blockIdx.x, o = threadIdx.x;
  float acc = b[o];
  #pragma unroll
  for (int k = 0; k < 5; ++k) acc = fmaf(ea[(size_t)e * 5 + k], w[k * H + o], acc);
  eh[(size_t)e * H + o] = fmaxf(acc, 0.f);
}

// W2bf[kk][i][o]: rows 0..63 = nn2_w (bf16), row 64 = nn2_b reshaped [i][o]
__global__ void k_w2cvt(const float* __restrict__ w2, const float* __restrict__ b2,
                        bf16* __restrict__ out) {
  int i = blockIdx.x * blockDim.x + threadIdx.x;
  if (i < KSL * 4096) {
    float v = (i < 64 * 4096) ? w2[i] : b2[i - 64 * 4096];
    out[i] = (bf16)v;
  }
}

// per-iteration: hbf = (bf16)h  -- staged once so k_tgemm can async-copy it
__global__ void k_hcvt(const float* __restrict__ h, bf16* __restrict__ hbf) {
  int i = blockIdx.x * blockDim.x + threadIdx.x;
  if (i < NPAD * H) hbf[i] = (bf16)h[i];
}

__global__ void k_deg(const long long* __restrict__ ei, float* __restrict__ deg) {
  int e = blockIdx.x * blockDim.x + threadIdx.x;
  if (e < N_EDGES) atomicAdd(&deg[(int)ei[N_EDGES + e]], 1.0f);
}

__global__ void k_invden(const float* __restrict__ deg, float* __restrict__ inv) {
  int n = blockIdx.x * blockDim.x + threadIdx.x;
  if (n < NPAD) inv[n] = 1.0f / fmaxf(deg[n], 1.0f);
}

// ------------------------------------------------ T[n,kk,o] = sum_i h[n,i]*W2[kk,i,o]
// 64-node x 64-out tile per (m-tile, kk). A tile staged via async global->LDS copy
// (ASYNCcnt), B tile transposed through registers. bf16 WMMA, f32 acc, bf16 store.
__global__ __launch_bounds__(128) void k_tgemm(const bf16* __restrict__ hbf,
                                               const bf16* __restrict__ w2,
                                               bf16* __restrict__ T) {
  __shared__ __align__(16) bf16 sA[64][SP];   // [m][i]
  __shared__ __align__(16) bf16 sB[64][SP];   // [o][i]  (transposed slice)
  const int mbase = blockIdx.x * 64;
  const int kk    = blockIdx.y;
  const int tid   = threadIdx.x;

  // --- A tile: 64 rows x 64 bf16 = 8KB, 512 x 16B chunks, async copy to LDS
  const uint32_t sAoff = (uint32_t)(uintptr_t)&sA[0][0];
  #pragma unroll
  for (int j = 0; j < 4; ++j) {
    int c = tid + j * 128;                 // chunk id 0..511
    int r = c >> 3, q = c & 7;             // row, 16B-chunk within row
    uint64_t ga = (uint64_t)(uintptr_t)(hbf + (size_t)(mbase + r) * H + q * 8);
    uint32_t la = sAoff + (uint32_t)((r * SP + q * 8) * 2);
    asm volatile("global_load_async_to_lds_b128 %0, %1, off"
                 :: "v"(la), "v"(ga) : "memory");
  }

  // --- B tile: transpose w2[kk][i][o] -> sB[o][i] through registers
  const bf16* wrow = w2 + (size_t)kk * 4096;   // [i][o]
  for (int idx = tid; idx < 4096; idx += 128) {
    sB[idx & 63][idx >> 6] = wrow[idx];
  }

  asm volatile("s_wait_asynccnt 0x0" ::: "memory");  // own wave's async copies done
  __syncthreads();                                   // publish LDS across waves

  const int wave  = tid >> 5;
  const int lane  = tid & 31;
  const int lrow  = lane & 15;
  const int ahalf = (lane & 16) ? 8 : 0;   // A: K interleave halves
  const int khalf = (lane & 16) ? 16 : 0;  // unused; B reads with ahalf, same k order as A
  const int m     = wave * 16 + lrow;

  v8f acc[4] = {v8f{}, v8f{}, v8f{}, v8f{}};
  #pragma unroll
  for (int s = 0; s < 2; ++s) {
    const int k0 = s * 32;
    bf16x8 al = *(const bf16x8*)&sA[m][k0 + ahalf];
    bf16x8 ah = *(const bf16x8*)&sA[m][k0 + 16 + ahalf];
    v16bf a;
    #pragma unroll
    for (int j = 0; j < 8; ++j) { a[j] = al[j]; a[8 + j] = ah[j]; }
    #pragma unroll
    for (int nt = 0; nt < 4; ++nt) {
      const int o = nt * 16 + lrow;
      bf16x8 bl = *(const bf16x8*)&sB[o][k0 + ahalf];
      bf16x8 bh = *(const bf16x8*)&sB[o][k0 + 16 + ahalf];
      v16bf b;
      #pragma unroll
      for (int j = 0; j < 8; ++j) { b[j] = bl[j]; b[8 + j] = bh[j]; }
      acc[nt] = __builtin_amdgcn_wmma_f32_16x16x32_bf16(
          false, a, false, b, (short)0, acc[nt], false, false);
    }
  }
  const int mc = wave * 16 + ((lane & 16) ? 8 : 0);
  #pragma unroll
  for (int nt = 0; nt < 4; ++nt) {
    const int o = nt * 16 + lrow;
    #pragma unroll
    for (int r = 0; r < 8; ++r) {
      T[((size_t)(mbase + mc + r) * KSL + kk) * 64 + o] = (bf16)acc[nt][r];
    }
  }
}

// ---------------- msg[e,o] = sum_kk eh[e,kk]*T[src,kk,o] + T[src,64,o]; agg[dst] += msg
__global__ __launch_bounds__(32) void k_edge(const long long* __restrict__ ei,
                                             const float* __restrict__ eh,
                                             const bf16* __restrict__ T,
                                             float* __restrict__ agg) {
  const int e    = blockIdx.x;
  const int lane = threadIdx.x;                 // 0..31, 2 outputs per lane
  const int src  = (int)ei[e];
  const int dst  = (int)ei[N_EDGES + e];
  const uint32_t* Tn = (const uint32_t*)(T + (size_t)src * KSL * 64);
  uint32_t u = Tn[64 * 32 + lane];              // bias slice kk=64
  float a0 = __uint_as_float(u << 16);
  float a1 = __uint_as_float(u & 0xffff0000u);
  const float* ehr = eh + (size_t)e * H;
  for (int kk = 0; kk < 64; ++kk) {
    float c = ehr[kk];                          // uniform across wave -> scalar load
    uint32_t t = Tn[kk * 32 + lane];            // 128B coalesced, L2-resident
    a0 = fmaf(c, __uint_as_float(t << 16), a0);
    a1 = fmaf(c, __uint_as_float(t & 0xffff0000u), a1);
  }
  float* arow = agg + (size_t)dst * H;
  atomicAdd(&arow[2 * lane], a0);
  atomicAdd(&arow[2 * lane + 1], a1);
}

// h_new = relu(h @ root_w + agg/denom + conv_b)
__global__ void k_update(const float* __restrict__ h, const float* __restrict__ rw,
                         const float* __restrict__ cb, const float* __restrict__ agg,
                         const float* __restrict__ inv, float* __restrict__ hn) {
  int n = blockIdx.x, o = threadIdx.x;
  float v = 0.f;
  if (n < N_NODES) {
    float acc = cb[o] + agg[(size_t)n * H + o] * inv[n];
    const float* hr = h + (size_t)n * H;
    for (int i = 0; i < H; ++i) acc = fmaf(hr[i], rw[i * H + o], acc);
    v = fmaxf(acc, 0.f);
  }
  hn[(size_t)n * H + o] = v;
}

__global__ void k_rowsum(const float* __restrict__ h, float* __restrict__ outg) {
  __shared__ float red[4][64];
  int o = threadIdx.x, y = threadIdx.y;
  int n0 = blockIdx.x * 64;
  float acc = 0.f;
  for (int n = n0 + y; n < n0 + 64 && n < N_NODES; n += 4)
    acc += h[(size_t)n * H + o];
  red[y][o] = acc;
  __syncthreads();
  if (y == 0) atomicAdd(&outg[o], red[0][o] + red[1][o] + red[2][o] + red[3][o]);
}

// 69 -> 384 -> (384x384)x6 -> 1, single workgroup
__global__ __launch_bounds__(384) void k_head(const float* __restrict__ outg,
    const float* __restrict__ vpa, const float* __restrict__ mz,
    const float* __restrict__ adduct,
    const float* __restrict__ bw, const float* __restrict__ bb,
    const float* __restrict__ l1w, const float* __restrict__ l1b,
    const float* __restrict__ l2w, const float* __restrict__ l2b,
    float* __restrict__ out) {
  __shared__ float vec[69];
  __shared__ float buf[384];
  int t = threadIdx.x;
  if (t < 64) vec[t] = outg[t];
  else if (t == 64) vec[64] = vpa[0];
  else if (t == 65) vec[65] = mz[0];
  else if (t < 69) vec[t] = adduct[t - 66];
  __syncthreads();
  float acc = bb[t];
  for (int k = 0; k < 69; ++k) acc = fmaf(vec[k], bw[k * 384 + t], acc);
  float cur = fmaxf(acc, 0.f);
  for (int l = 0; l < 6; ++l) {
    buf[t] = cur;
    __syncthreads();
    acc = l1b[t];
    for (int k = 0; k < 384; ++k) acc = fmaf(buf[k], l1w[k * 384 + t], acc);
    cur = fmaxf(acc, 0.f);
    __syncthreads();
  }
  buf[t] = cur * l2w[t];
  __syncthreads();
  if (t == 0) {
    float s = l2b[0];
    for (int k = 0; k < 384; ++k) s += buf[k];
    out[0] = s;
  }
}

extern "C" void kernel_launch(void* const* d_in, const int* in_sizes, int n_in,
                              void* d_out, int out_size, void* d_ws, size_t ws_size,
                              hipStream_t stream) {
  (void)in_sizes; (void)n_in; (void)out_size; (void)ws_size;
  const float*     x      = (const float*)d_in[0];
  const long long* ei     = (const long long*)d_in[1];   // int64 [2, E]
  const float*     ea     = (const float*)d_in[2];
  const float*     vpa    = (const float*)d_in[3];
  const float*     mz     = (const float*)d_in[4];
  const float*     adduct = (const float*)d_in[5];
  const float*     l0w    = (const float*)d_in[6];
  const float*     l0b    = (const float*)d_in[7];
  const float*     n1w    = (const float*)d_in[8];
  const float*     n1b    = (const float*)d_in[9];
  const float*     n2w    = (const float*)d_in[10];
  const float*     n2b    = (const float*)d_in[11];
  const float*     rw     = (const float*)d_in[12];
  const float*     cb     = (const float*)d_in[13];
  const float*     bw     = (const float*)d_in[14];
  const float*     bb     = (const float*)d_in[15];
  const float*     l1w    = (const float*)d_in[16];
  const float*     l1b    = (const float*)d_in[17];
  const float*     l2w    = (const float*)d_in[18];
  const float*     l2b    = (const float*)d_in[19];

  char* ws = (char*)d_ws;
  size_t off = 0;
  auto alloc = [&](size_t bytes) -> void* {
    void* p = ws + off;
    off = (off + bytes + 255) & ~(size_t)255;
    return p;
  };
  float* hA   = (float*)alloc((size_t)NPAD * H * 4);
  float* hB   = (float*)alloc((size_t)NPAD * H * 4);
  bf16*  hbf  = (bf16*)alloc((size_t)NPAD * H * 2);
  float* ehid = (float*)alloc((size_t)N_EDGES * H * 4);
  bf16*  w2b  = (bf16*)alloc((size_t)KSL * 4096 * 2);
  bf16*  T    = (bf16*)alloc((size_t)NPAD * KSL * 64 * 2);   // 84 MB, L2-resident
  float* agg  = (float*)alloc((size_t)NPAD * H * 4);
  float* deg  = (float*)alloc((size_t)NPAD * 4);
  float* inv  = (float*)alloc((size_t)NPAD * 4);
  float* outg = (float*)alloc(64 * 4);

  k_lin0<<<NPAD, 64, 0, stream>>>(x, l0w, l0b, hA);
  k_nn1<<<N_EDGES, 64, 0, stream>>>(ea, n1w, n1b, ehid);
  k_w2cvt<<<(KSL * 4096 + 255) / 256, 256, 0, stream>>>(n2w, n2b, w2b);
  k_zero<<<(NPAD + 255) / 256, 256, 0, stream>>>(deg, NPAD);
  k_deg<<<(N_EDGES + 255) / 256, 256, 0, stream>>>(ei, deg);
  k_invden<<<(NPAD + 255) / 256, 256, 0, stream>>>(deg, inv);

  float* hc = hA;
  float* hn = hB;
  for (int it = 0; it < 3; ++it) {
    k_hcvt<<<(NPAD * H + 255) / 256, 256, 0, stream>>>(hc, hbf);
    k_tgemm<<<dim3(MT, KSL), 128, 0, stream>>>(hbf, w2b, T);
    k_zero<<<(NPAD * H + 255) / 256, 256, 0, stream>>>(agg, NPAD * H);
    k_edge<<<N_EDGES, 32, 0, stream>>>(ei, ehid, T, agg);
    k_update<<<NPAD, 64, 0, stream>>>(hc, rw, cb, agg, inv, hn);
    float* tmp = hc; hc = hn; hn = tmp;
  }

  k_zero<<<1, 64, 0, stream>>>(outg, 64);
  k_rowsum<<<MT, dim3(64, 4), 0, stream>>>(hc, outg);
  k_head<<<1, 384, 0, stream>>>(outg, vpa, mz, adduct, bw, bb, l1w, l1b,
                                l2w, l2b, (float*)d_out);
}
